// HyperGCN_66975720014006
// MI455X (gfx1250) — matrix operand order repaired and checked by reading, not compiled, not run on hardware
//
#include <hip/hip_runtime.h>

// ---------------- problem constants ----------------
#define BD    128          // dialogues
#define LL    110          // length
#define DD    512          // feature dim (= NH)
#define NROW  (BD*LL)      // 14080 rows of a/v/l
#define NNODE (3*BD*LL)    // 42240 hypergraph nodes
#define OUTW  3072         // output row width

typedef __attribute__((ext_vector_type(16))) __bf16 v16bf;
typedef __attribute__((ext_vector_type(8)))  float  v8f;

union FragU { uint4 u[2]; v16bf v; };

// ---------------- bf16 helpers (RNE) ----------------
__device__ __forceinline__ unsigned short f2bf(float f) {
    unsigned u = __float_as_uint(f);
    u += 0x7FFFu + ((u >> 16) & 1u);
    return (unsigned short)(u >> 16);
}
__device__ __forceinline__ float bf2f(unsigned short h) {
    return __uint_as_float(((unsigned)h) << 16);
}

// ---------------- fill ----------------
__global__ void fill_kernel(float* __restrict__ p, int count, float val) {
    int i = blockIdx.x * blockDim.x + threadIdx.x;
    int s = gridDim.x * blockDim.x;
    for (; i < count; i += s) p[i] = val;
}

// ---------------- fp32 -> (bf16 hi, bf16 lo) ----------------
__global__ void convert_pair_kernel(const float* __restrict__ src,
                                    unsigned short* __restrict__ dh,
                                    unsigned short* __restrict__ dl, int count) {
    int i = blockIdx.x * blockDim.x + threadIdx.x;
    int s = gridDim.x * blockDim.x;
    for (; i < count; i += s) {
        float v = src[i];
        unsigned short h = f2bf(v);
        dh[i] = h;
        dl[i] = f2bf(v - bf2f(h));
    }
}

// ---------------- x *= 1/Dnode (optional), then bf16 pair ----------------
__global__ void scale_convert_kernel(float* __restrict__ x,
                                     const float* __restrict__ dnode,
                                     unsigned short* __restrict__ dh,
                                     unsigned short* __restrict__ dl,
                                     int count, int use_di) {
    int i = blockIdx.x * blockDim.x + threadIdx.x;
    int s = gridDim.x * blockDim.x;
    for (; i < count; i += s) {
        float v = x[i];
        if (use_di) {
            float dn = dnode[i >> 9];               // row = i / 512
            v *= (dn > 0.f ? 1.f / dn : 1.f);
            x[i] = v;
        }
        unsigned short h = f2bf(v);
        dh[i] = h;
        dl[i] = f2bf(v - bf2f(h));
    }
}

// ---------------- build feats: l + spk_emb | a | v ; write out + bf16 pair --
__global__ __launch_bounds__(256)
void build_feats_kernel(const float* __restrict__ a, const float* __restrict__ v,
                        const float* __restrict__ l, const float* __restrict__ qmask,
                        const float* __restrict__ spk_emb,
                        float* __restrict__ out,
                        unsigned short* __restrict__ Ah, unsigned short* __restrict__ Al) {
    int n   = blockIdx.x;                 // node id 0..NNODE-1
    int d   = n / (3 * LL);
    int rem = n - d * 3 * LL;
    int blk = rem / LL;                   // 0:l 1:a 2:v
    int i   = rem - blk * LL;
    int src = d * LL + i;
    const float* srow = (blk == 0 ? l : blk == 1 ? a : v) + (size_t)src * DD;
    const float* addrow = nullptr;
    if (blk == 0) {
        int qb = (i * BD + d) * 2;        // qmask is (LL, BD, 2)
        int spk = (qmask[qb + 1] > qmask[qb + 0]) ? 1 : 0;
        addrow = spk_emb + spk * DD;
    }
    size_t obase = (size_t)(d * LL + i) * OUTW + blk * 1024;
    size_t nbase = (size_t)n * DD;
    for (int c = threadIdx.x; c < DD; c += blockDim.x) {
        float val = srow[c] + (addrow ? addrow[c] : 0.f);
        out[obase + c] = val;
        unsigned short h = f2bf(val);
        Ah[nbase + c] = h;
        Al[nbase + c] = f2bf(val - bf2f(h));
    }
}

// ---------------- Dnode[n] += hw[he_edge[j]] ----------------
__global__ void dnode_kernel(float* __restrict__ dnode,
                             const int* __restrict__ he_node,
                             const int* __restrict__ he_edge,
                             const float* __restrict__ hw, int nnz) {
    int j = blockIdx.x * blockDim.x + threadIdx.x;
    int s = gridDim.x * blockDim.x;
    for (; j < nnz; j += s)
        atomicAdd(&dnode[he_node[j]], hw[he_edge[j]]);
}

// ---------------- split-bf16 WMMA GEMM: C = A(hi+lo) @ W(hi+lo)^T + bias ----
// A: M x 512 (bf16 pair, row-major), W: 512 x 512 row-major [n][k] (bf16 pair)
// wave computes a 16x64 strip; block (8 waves) covers 16 x 512.
__global__ __launch_bounds__(256)
void gemm_bf16x2_kernel(const unsigned short* __restrict__ Ah,
                        const unsigned short* __restrict__ Al,
                        const unsigned short* __restrict__ Wh,
                        const unsigned short* __restrict__ Wl,
                        const float* __restrict__ bias,
                        float* __restrict__ C) {
    const int wave = threadIdx.x >> 5;
    const int lane = threadIdx.x & 31;
    const int m0   = blockIdx.x * 16;
    const int n0   = wave * 64;

    const int arow = m0 + (lane & 15);
    const int asel = (lane >> 4) * 8;                 // K sub-offset for A
    const unsigned short* aph = Ah + (size_t)arow * DD;
    const unsigned short* apl = Al + (size_t)arow * DD;

    const int bn   = lane & 15;                       // B column (N)
    const int bsel = (lane >> 4) * 8;                 // K sub-offset for B

    v8f acc[4] = {};

    for (int k0 = 0; k0 < DD; k0 += 32) {
        FragU fah, fal;
        fah.u[0] = *(const uint4*)(aph + k0 + asel);
        fah.u[1] = *(const uint4*)(aph + k0 + asel + 16);
        fal.u[0] = *(const uint4*)(apl + k0 + asel);
        fal.u[1] = *(const uint4*)(apl + k0 + asel + 16);
#pragma unroll
        for (int t = 0; t < 4; ++t) {
            const size_t boff = (size_t)(n0 + t * 16 + bn) * DD + k0 + bsel;
            FragU fbh, fbl;
            fbh.u[0] = *(const uint4*)(Wh + boff);
            fbh.u[1] = *(const uint4*)(Wh + boff + 16);
            fbl.u[0] = *(const uint4*)(Wl + boff);
            fbl.u[1] = *(const uint4*)(Wl + boff + 16);
            acc[t] = __builtin_amdgcn_wmma_f32_16x16x32_bf16(
                false, fah.v, false, fbh.v, (short)0, acc[t], false, false);
            acc[t] = __builtin_amdgcn_wmma_f32_16x16x32_bf16(
                false, fah.v, false, fbl.v, (short)0, acc[t], false, false);
            acc[t] = __builtin_amdgcn_wmma_f32_16x16x32_bf16(
                false, fal.v, false, fbh.v, (short)0, acc[t], false, false);
        }
    }
    // C/D layout: VGPR r -> row m0 + r + (lane>>4)*8, col n0 + t*16 + (lane&15)
    const int rbase = m0 + (lane >> 4) * 8;
#pragma unroll
    for (int t = 0; t < 4; ++t) {
        const int n = n0 + t * 16 + (lane & 15);
        const float b = bias[n];
#pragma unroll
        for (int r = 0; r < 8; ++r)
            C[(size_t)(rbase + r) * DD + n] = acc[t][r] + b;
    }
}

// ---------------- per-edge segment sum (he_edge is sorted) ----------------
// em[e] = Bi * sum_j ew[j]*xt[he_node[j]]  + edge_attr(e)
__global__ __launch_bounds__(256)
void edge_em_kernel(const float* __restrict__ xt, float* __restrict__ em,
                    const int* __restrict__ he_node, const int* __restrict__ he_edge,
                    const float* __restrict__ ew, const int* __restrict__ he_type,
                    const float* __restrict__ attr1, const float* __restrict__ attr2,
                    int nnz) {
    const int e = blockIdx.x;
    int lo = 0, hi = nnz;
    while (lo < hi) { int m = (lo + hi) >> 1; if (he_edge[m] <  e) lo = m + 1; else hi = m; }
    const int start = lo;
    hi = nnz;
    while (lo < hi) { int m = (lo + hi) >> 1; if (he_edge[m] <= e) lo = m + 1; else hi = m; }
    const int end = lo;
    const float cnt = (float)(end - start);
    const float Bi  = cnt > 0.f ? 1.f / cnt : 1.f;
    const float tf  = (float)he_type[e];

    const int t = threadIdx.x;
    float acc0 = 0.f, acc1 = 0.f;
    for (int j = start; j < end; ++j) {
        const float c = ew[j];
        const float* row = xt + (size_t)he_node[j] * DD;
        acc0 += c * row[t];
        acc1 += c * row[t + 256];
    }
    em[(size_t)e * DD + t]       = acc0 * Bi + attr1[t] * tf       + attr2[t] * (1.f - tf);
    em[(size_t)e * DD + t + 256] = acc1 * Bi + attr1[t + 256] * tf + attr2[t + 256] * (1.f - tf);
}

// ---------------- node scatter: x[n] += hw[e] * em[e] ----------------
__global__ __launch_bounds__(256)
void node_scatter_kernel(const float* __restrict__ em, float* __restrict__ x,
                         const int* __restrict__ he_node, const int* __restrict__ he_edge,
                         const float* __restrict__ hw) {
    const int j = blockIdx.x;
    const int n = he_node[j];
    const int e = he_edge[j];
    const float wj = hw[e];
    const int t = threadIdx.x;
    atomicAdd(&x[(size_t)n * DD + t],       wj * em[(size_t)e * DD + t]);
    atomicAdd(&x[(size_t)n * DD + t + 256], wj * em[(size_t)e * DD + t + 256]);
}

// ---------------- write x half of the output ----------------
__global__ __launch_bounds__(256)
void final_out_kernel(const float* __restrict__ x, float* __restrict__ out) {
    int n   = blockIdx.x;
    int d   = n / (3 * LL);
    int rem = n - d * 3 * LL;
    int blk = rem / LL;
    int i   = rem - blk * LL;
    size_t obase = (size_t)(d * LL + i) * OUTW + blk * 1024 + 512;
    size_t nbase = (size_t)n * DD;
    for (int c = threadIdx.x; c < DD; c += blockDim.x)
        out[obase + c] = x[nbase + c];
}

// =====================================================================
extern "C" void kernel_launch(void* const* d_in, const int* in_sizes, int n_in,
                              void* d_out, int out_size, void* d_ws, size_t ws_size,
                              hipStream_t stream) {
    const float* a       = (const float*)d_in[0];
    const float* v       = (const float*)d_in[1];
    const float* l       = (const float*)d_in[2];
    const float* qmask   = (const float*)d_in[3];
    const float* spk_emb = (const float*)d_in[4];
    const float* fc1_w   = (const float*)d_in[5];
    const float* fc1_b   = (const float*)d_in[6];
    const float* hw      = (const float*)d_in[7];
    const float* ew      = (const float*)d_in[8];
    const float* attr1   = (const float*)d_in[9];
    const float* attr2   = (const float*)d_in[10];
    const float* conv_w  = (const float*)d_in[11];
    const float* conv_b  = (const float*)d_in[12];
    const int* he_node   = (const int*)d_in[13];
    const int* he_edge   = (const int*)d_in[14];
    const int* he_type   = (const int*)d_in[15];
    const int nnz = in_sizes[13];
    const int E   = in_sizes[15];
    float* out = (float*)d_out;

    // -------- workspace partition --------
    size_t off = 0;
    auto alloc = [&](size_t bytes) -> void* {
        void* p = (char*)d_ws + off;
        off = (off + bytes + 255) & ~(size_t)255;
        return p;
    };
    const size_t XE = (size_t)NNODE * DD;          // 21,626,880 elems
    unsigned short* Ah = (unsigned short*)alloc(XE * 2);
    unsigned short* Al = (unsigned short*)alloc(XE * 2);
    float* buf1  = (float*)alloc(XE * 4);          // xt
    float* buf2  = (float*)alloc(XE * 4);          // x
    float* em    = (float*)alloc((size_t)E * DD * 4);
    unsigned short* Wh = (unsigned short*)alloc((size_t)4 * DD * DD * 2);
    unsigned short* Wl = (unsigned short*)alloc((size_t)4 * DD * DD * 2);
    float* Dnode = (float*)alloc((size_t)NNODE * 4);

    const int WSZ = DD * DD;                       // 262144

    // weights -> bf16 hi/lo (slot 0: fc1, slots 1..3: conv layers)
    convert_pair_kernel<<<1024, 256, 0, stream>>>(fc1_w, Wh, Wl, WSZ);
    convert_pair_kernel<<<3072, 256, 0, stream>>>(conv_w, Wh + WSZ, Wl + WSZ, 3 * WSZ);

    // feats: write into output + bf16 pair for GEMM
    build_feats_kernel<<<NNODE, 256, 0, stream>>>(a, v, l, qmask, spk_emb, out, Ah, Al);

    // Dnode degrees
    fill_kernel<<<256, 256, 0, stream>>>(Dnode, NNODE, 0.f);
    dnode_kernel<<<330, 256, 0, stream>>>(Dnode, he_node, he_edge, hw, nnz);

    // x0 = feats @ fc1_w^T + fc1_b
    gemm_bf16x2_kernel<<<NNODE / 16, 256, 0, stream>>>(Ah, Al, Wh, Wl, fc1_b, buf2);
    scale_convert_kernel<<<8192, 256, 0, stream>>>(buf2, Dnode, Ah, Al, (int)XE, 0);

    for (int k = 0; k < 3; ++k) {
        // xt = x @ cw[k]^T + cb[k]
        gemm_bf16x2_kernel<<<NNODE / 16, 256, 0, stream>>>(
            Ah, Al, Wh + (size_t)(k + 1) * WSZ, Wl + (size_t)(k + 1) * WSZ,
            conv_b + k * DD, buf1);
        // em = Bi * segsum_edge(ew * xt[node]) + edge_attr
        edge_em_kernel<<<E, 256, 0, stream>>>(buf1, em, he_node, he_edge, ew,
                                              he_type, attr1, attr2, nnz);
        // x = Di * segsum_node(hw[edge] * em[edge])
        fill_kernel<<<8192, 256, 0, stream>>>(buf2, (int)XE, 0.f);
        node_scatter_kernel<<<nnz, 256, 0, stream>>>(em, buf2, he_node, he_edge, hw);
        scale_convert_kernel<<<8192, 256, 0, stream>>>(buf2, Dnode, Ah, Al, (int)XE, 1);
    }

    // write x half of output
    final_out_kernel<<<NNODE, 256, 0, stream>>>(buf2, out);
}
